// GAT_85590108274943
// MI455X (gfx1250) — matrix operand order repaired and checked by reading, not compiled, not run on hardware
//
#include <hip/hip_runtime.h>
#include <stdint.h>

// GAT on MI455X (gfx1250): flash-attention-style fused masked softmax + WMMA,
// async direct-to-LDS double-buffered conv tiles, wave-local (barrier-free) LDS.
// Reference shapes: N=8192, D_IN=256, F=64, H=2, f32 in / f32 out.

#define N_NODES 8192
#define D_IN    256
#define F_DIM   64
#define H_HEADS 2
#define NEG_BIG (-1e10f)

typedef __attribute__((ext_vector_type(16))) _Float16 v16h;
typedef __attribute__((ext_vector_type(8)))  float    v8f;
typedef __attribute__((ext_vector_type(2)))  _Float16 h2;

// ---- CDNA5 async global->LDS path (guarded; falls back to sync staging) ----
#if __has_builtin(__builtin_amdgcn_global_load_async_to_lds_b128)
#define HAVE_ASYNC_LDS 1
// builtin expects: (v4i addrspace(1)*, v4i addrspace(3)*, i32 offset, i32 cpol)
typedef int v4i_ __attribute__((vector_size(16)));
typedef __attribute__((address_space(1))) v4i_* as1_b128_t;
typedef __attribute__((address_space(3))) v4i_* as3_b128_t;
#define TO_GLOBAL_B128(p) ((as1_b128_t)(uintptr_t)(p))
#define TO_LDS_B128(p)    ((as3_b128_t)(uint32_t)(uintptr_t)(p))
#else
#define HAVE_ASYNC_LDS 0
#endif

#if HAVE_ASYNC_LDS
#if __has_builtin(__builtin_amdgcn_s_wait_asynccnt)
#define WAIT_ASYNC_LE(n) __builtin_amdgcn_s_wait_asynccnt(n)
#else
#define WAIT_ASYNC_LE(n) asm volatile("s_wait_asynccnt %0" ::"n"(n) : "memory")
#endif
#else
#define WAIT_ASYNC_LE(n) do { } while (0)
#endif

// Intra-wave LDS producer->consumer needs no s_barrier: DS ops from one wave
// complete in order. Just stop the compiler from reordering across the point.
__device__ __forceinline__ void lds_fence() {
  __builtin_amdgcn_wave_barrier();
  asm volatile("" ::: "memory");
}

// ---------------------------------------------------------------------------
// Kernel 1: conv = x @ W + b (f32), a_self/a_neigh (f32), convT stored f16
// transposed [H][F][N] so the WMMA B operand's (K,K+1) pairs are contiguous.
// ---------------------------------------------------------------------------
__global__ __launch_bounds__(256) void gat_prep(
    const float* __restrict__ x, const float* __restrict__ kernels,
    const float* __restrict__ kbias, const float* __restrict__ attk,
    const float* __restrict__ attb,
    _Float16* __restrict__ convT, float* __restrict__ a_self,
    float* __restrict__ a_neigh) {
  int t = blockIdx.x * blockDim.x + threadIdx.x;   // 0 .. H*N-1
  int h = t >> 13;                                 // N = 8192 = 2^13
  int n = t & (N_NODES - 1);

  float c[F_DIM];
#pragma unroll
  for (int f = 0; f < F_DIM; ++f) c[f] = kbias[h * F_DIM + f];

  const float* xr = x + (size_t)n * D_IN;
  const float* W  = kernels + (size_t)h * D_IN * F_DIM;
  for (int d = 0; d < D_IN; ++d) {
    float xv = xr[d];
#pragma unroll
    for (int f = 0; f < F_DIM; ++f)
      c[f] = fmaf(xv, W[d * F_DIM + f], c[f]);     // W uniform per wave -> scalar loads
  }

  float s = attb[h], g = 0.f;
#pragma unroll
  for (int f = 0; f < F_DIM; ++f) {
    s = fmaf(c[f], attk[h * 2 * F_DIM + f], s);
    g = fmaf(c[f], attk[h * 2 * F_DIM + F_DIM + f], g);
    convT[((size_t)(h * F_DIM + f)) * N_NODES + n] = (_Float16)c[f];  // coalesced in n
  }
  a_self[h * N_NODES + n]  = s;
  a_neigh[h * N_NODES + n] = g;
}

// ---------------------------------------------------------------------------
// Stage convT[f][j0..j0+31] (64 rows x 64B = 4KB) into an LDS buffer.
// Async path: 8x global_load_async_to_lds_b128 (512B per instruction).
// ---------------------------------------------------------------------------
__device__ __forceinline__ void stage_conv(const _Float16* __restrict__ convTh,
                                           _Float16* lbuf, int j0, int lane) {
#pragma unroll
  for (int t = 0; t < 8; ++t) {
    int q = t * 32 + lane;                          // 0..255 16B packets
    int f = q >> 2;
    int u = q & 3;
    const _Float16* gp = convTh + (size_t)f * N_NODES + j0 + u * 8;
#if HAVE_ASYNC_LDS
    __builtin_amdgcn_global_load_async_to_lds_b128(TO_GLOBAL_B128(gp),
                                                   TO_LDS_B128(lbuf + q * 8),
                                                   0, 0);
#else
    ((uint4*)lbuf)[q] = *(const uint4*)gp;
#endif
  }
}

// ---------------------------------------------------------------------------
// Kernel 2: fused masked-softmax attention + attn@conv via v_wmma_f32_16x16x32_f16.
// Block = 64 threads = 2 waves; wave == head. Waves are LDS-independent but are
// loosely re-synced every 8 chunks so both heads consume the same adj cache
// lines while WGP$/L2-resident (adj = 256MB, the HBM roofline term).
// ---------------------------------------------------------------------------
__global__ __launch_bounds__(64) void gat_attn(
    const float* __restrict__ adj, const _Float16* __restrict__ convT,
    const float* __restrict__ a_self_g, const float* __restrict__ a_neigh_g,
    float* __restrict__ out) {
  __shared__ float    sc_s[H_HEADS][16][32];           // score tile (i x j)
  __shared__ _Float16 cv_s[H_HEADS][2][F_DIM][32];     // double-buffered conv tile
  __shared__ float    row_s[H_HEADS][16];              // per-row broadcast

  const int h    = threadIdx.x >> 5;                   // wave id == head
  const int lane = threadIdx.x & 31;
  const int m    = lane & 15;
  const int g    = lane >> 4;
  const int i0   = blockIdx.x * 16;

  float asf[16];
#pragma unroll
  for (int r = 0; r < 16; ++r) asf[r] = a_self_g[h * N_NODES + i0 + r];

  float m_run = -INFINITY;
  float l_run = 0.f;
  v8f acc[4];
  const v8f vzero = {0.f, 0.f, 0.f, 0.f, 0.f, 0.f, 0.f, 0.f};
#pragma unroll
  for (int fb = 0; fb < 4; ++fb) acc[fb] = vzero;

  const _Float16* convTh = convT + (size_t)h * F_DIM * N_NODES;
  const float*    aneigh = a_neigh_g + h * N_NODES;

  const int NCHUNK = N_NODES / 32;
  stage_conv(convTh, &cv_s[h][0][0][0], 0, lane);      // prime buffer 0

  for (int c = 0; c < NCHUNK; ++c) {
    const int j0  = c * 32;
    const int cur = c & 1;
    const bool has_next = (c + 1) < NCHUNK;
    if (has_next)                                      // prefetch next tile
      stage_conv(convTh, &cv_s[h][cur ^ 1][0][0], j0 + 32, lane);

    // ---- scores: lane = column j0+lane; adj loads fully coalesced.
    // Mask select is bit-exact to the reference's exp(adj*-1e10)*-1e10 trick
    // for binary adj (adj=1 -> exp underflows to 0 -> -0.0; adj=0 -> -1e10),
    // and saves one v_exp + two v_mul per element.
    float an = aneigh[j0 + lane];
    const float* adjp = adj + (size_t)i0 * N_NODES + j0 + lane;
#pragma unroll
    for (int r = 0; r < 16; ++r) {
      float a = adjp[(size_t)r * N_NODES];
      float e = asf[r] + an;
      e = (e > 0.f) ? e : (__expf(e) - 1.f);           // elu, alpha=1
      float mask = (a == 0.f) ? NEG_BIG : -0.0f;
      sc_s[h][r][lane] = e + mask;
    }
    lds_fence();

    // ---- gather scores per 16-bit A-matrix layout (lane: M=m, K pairs/VGPR)
    float s0[8], s1[8];
    float cmax = -INFINITY;
#pragma unroll
    for (int v = 0; v < 8; ++v) {
      int k = ((v < 4) ? 0 : 16) + g * 8 + (v & 3) * 2;
      float2 p = *(const float2*)&sc_s[h][m][k];
      s0[v] = p.x; s1[v] = p.y;
      cmax = fmaxf(cmax, fmaxf(p.x, p.y));
    }
    cmax = fmaxf(cmax, __shfl_xor(cmax, 16, 32));      // row max across lane halves
    float m_new  = fmaxf(m_run, cmax);
    float rescal = __expf(m_run - m_new);              // first iter: exp(-inf)=0
    m_run = m_new;

    // ---- rescale accumulators only when some row max moved (wave-uniform)
    if (__any(rescal != 1.0f)) {
      if (lane < 16) row_s[h][lane] = rescal;
      lds_fence();
      float scl[8];
#pragma unroll
      for (int r = 0; r < 8; ++r) scl[r] = row_s[h][r + 8 * g];
#pragma unroll
      for (int fb = 0; fb < 4; ++fb)
#pragma unroll
        for (int r = 0; r < 8; ++r) acc[fb][r] *= scl[r];
    }

    // ---- P = exp(score - m_new) packed f16 into A; row-sum for l
    v16h A;
    float rsum = 0.f;
#pragma unroll
    for (int v = 0; v < 8; ++v) {
      float p0 = __expf(s0[v] - m_new);
      float p1 = __expf(s1[v] - m_new);
      rsum += p0 + p1;
      A[2 * v]     = (_Float16)p0;
      A[2 * v + 1] = (_Float16)p1;
    }
    rsum += __shfl_xor(rsum, 16, 32);
    l_run = l_run * rescal + rsum;

    // ---- current conv tile must be resident (async loads complete in order:
    //      <=8 outstanding means the prefetch is the only one in flight)
    if (has_next) { WAIT_ASYNC_LE(8); } else { WAIT_ASYNC_LE(0); }
    lds_fence();

    // ---- B per 32x16 f16 layout (lane: N=m, K same order as A) + WMMA
#pragma unroll
    for (int fb = 0; fb < 4; ++fb) {
      v16h B;
      int col = fb * 16 + m;
#pragma unroll
      for (int v = 0; v < 8; ++v) {
        h2 p = *(const h2*)&cv_s[h][cur][col][((v < 4) ? 0 : 16) + g * 8 + (v & 3) * 2];  // contiguous b32
        B[2 * v]     = p.x;
        B[2 * v + 1] = p.y;
      }
      acc[fb] = __builtin_amdgcn_wmma_f32_16x16x32_f16(
          false, A, false, B, (short)0, acc[fb], false, false);
    }
    lds_fence();   // before next chunk overwrites sc_s / this cv_s buffer

    // keep the two head-waves loosely in lockstep for adj cache-line sharing
    if ((c & 7) == 7) __syncthreads();
  }

  // ---- finalize: out[i, h*F + f] = acc / l_run(row)
  if (lane < 16) row_s[h][lane] = l_run;
  lds_fence();
  float linv[8];
#pragma unroll
  for (int r = 0; r < 8; ++r) linv[r] = 1.f / row_s[h][r + 8 * g];
#pragma unroll
  for (int fb = 0; fb < 4; ++fb) {
#pragma unroll
    for (int r = 0; r < 8; ++r) {
      int row = i0 + r + 8 * g;                        // C layout: M = r + 8*(lane/16)
      int col = h * F_DIM + fb * 16 + m;               // N = lane%16
      out[(size_t)row * (H_HEADS * F_DIM) + col] = acc[fb][r] * linv[r];
    }
  }
}

// ---------------------------------------------------------------------------
extern "C" void kernel_launch(void* const* d_in, const int* in_sizes, int n_in,
                              void* d_out, int out_size, void* d_ws, size_t ws_size,
                              hipStream_t stream) {
  const float* x       = (const float*)d_in[0];
  const float* adj     = (const float*)d_in[1];
  const float* kernels = (const float*)d_in[2];
  const float* kbias   = (const float*)d_in[3];
  const float* attk    = (const float*)d_in[4];
  const float* attb    = (const float*)d_in[5];
  float* out = (float*)d_out;

  // workspace: convT f16 [H][F][N] (2MB) | a_self f32 [H][N] | a_neigh f32 [H][N]
  _Float16* convT = (_Float16*)d_ws;
  float* a_self  = (float*)((char*)d_ws +
                   (size_t)H_HEADS * F_DIM * N_NODES * sizeof(_Float16));
  float* a_neigh = a_self + H_HEADS * N_NODES;

  gat_prep<<<(H_HEADS * N_NODES) / 256, 256, 0, stream>>>(
      x, kernels, kbias, attk, attb, convT, a_self, a_neigh);

  gat_attn<<<N_NODES / 16, 64, 0, stream>>>(adj, convT, a_self, a_neigh, out);
}
